// PriorLayer_45286135169109
// MI455X (gfx1250) — matrix operand order repaired and checked by reading, not compiled, not run on hardware
//
#include <hip/hip_runtime.h>

#define DIM 256
#define SEQ 65536
#define CHUNK_LEN 256
#define WARMUP 64
#define NCHUNKS (SEQ / CHUNK_LEN)
#define PF 8

typedef __attribute__((ext_vector_type(16))) __bf16 v16bf;
typedef __attribute__((ext_vector_type(8)))  float  v8f;

__device__ __forceinline__ float wave_sum(float x) {
#pragma unroll
  for (int off = 16; off > 0; off >>= 1)
    x += __shfl_xor(x, off, 32);
  return x;
}

// Keep a bf16x16 WMMA operand pinned in VGPRs across a scheduling point.
__device__ __forceinline__ void pin_operand(v16bf& b) {
  v8f t = __builtin_bit_cast(v8f, b);
  asm volatile("" : "+v"(t));
  b = __builtin_bit_cast(v16bf, t);
}

__global__ __launch_bounds__(256, 1)
void prior_layer_kernel(const float* __restrict__ probs,
                        const float* __restrict__ T,
                        float* __restrict__ outP,
                        float* __restrict__ outU) {
  __shared__ alignas(32) __bf16 u_sh[DIM];   // belief vector (bf16, WMMA B source)
  __shared__ float pp_sh[DIM];               // p_prior = T @ u (f32)
  __shared__ float sacc[2];                  // normalizer accumulators (double-buffered)
  __shared__ float eacc[2];                  // entropy accumulators

  const int tid  = threadIdx.x;
  const int wave = tid >> 5;                 // 0..7
  const int lane = tid & 31;
  const int half = lane >> 4;                // 0/1

  // --- one-time: load T rows into registers in WMMA 16-bit A-operand layout ---
  // A 16x32 bf16, contiguous K as chosen here (B matches): lanes 0-15 = rows, slots 0..15 -> K {0..15};
  // lanes 16-31: same rows, K {16..31}.
  v16bf Areg[2][8];
#pragma unroll
  for (int j = 0; j < 2; ++j) {
    const int row = (wave * 2 + j) * 16 + (lane & 15);
    const float* trow = T + row * DIM;
#pragma unroll
    for (int k = 0; k < 8; ++k) {
      const int k0 = k * 32 + half * 16;      // slots 0..7
      const int k1 = k * 32 + half * 16 + 8;  // slots 8..15
      v16bf a;
#pragma unroll
      for (int s = 0; s < 8; ++s) {
        a[s]     = (__bf16)trow[k0 + s];
        a[s + 8] = (__bf16)trow[k1 + s];
      }
      Areg[j][k] = a;
    }
  }

  // initial belief: uniform
  u_sh[tid] = (__bf16)(1.0f / (float)DIM);
  if (tid < 2) { sacc[tid] = 0.0f; eacc[tid] = 0.0f; }
  __syncthreads();

  const int chunk   = blockIdx.x;
  const int t_write = chunk * CHUNK_LEN;
  const int t_begin = (chunk == 0) ? 0 : (t_write - WARMUP);
  const int t_end   = t_write + CHUNK_LEN;

  const v16bf* uvec = (const v16bf*)u_sh;     // [k*2 + half] = this half's K segment
  const int m0 = wave * 32;                   // this wave owns output rows m0..m0+31

  for (int t = t_begin; t < t_end; ++t) {
    // observation row (kept hot via prefetch issued PF steps earlier)
    float obs = probs[t * DIM + tid];
    if (t + PF < SEQ) __builtin_prefetch(probs + (t + PF) * DIM + tid, 0, 3);

    // ---- issue ALL B loads up front and pin them live: one dscnt wait,
    //      then an uninterrupted 16-WMMA burst ----
    v16bf bv[8];
#pragma unroll
    for (int k = 0; k < 8; ++k) bv[k] = uvec[k * 2 + half];
#pragma unroll
    for (int k = 0; k < 8; ++k) pin_operand(bv[k]);

    // ---- matvec p_prior = T @ u via WMMA (u broadcast into all 16 B columns) ----
    // Two D->C accumulation chains (full-rate; no WMMA->WMMA A/B hazard).
    v8f acc0 = {}, acc1 = {};
#pragma unroll
    for (int k = 0; k < 8; ++k) {
      acc0 = __builtin_amdgcn_wmma_f32_16x16x32_bf16(false, Areg[0][k], false, bv[k],
                                                     (short)0, acc0, false, false);
      acc1 = __builtin_amdgcn_wmma_f32_16x16x32_bf16(false, Areg[1][k], false, bv[k],
                                                     (short)0, acc1, false, false);
    }

    // D layout: VGPR r holds row (16*mt + r) in lanes 0-15 and row (16*mt + 8 + r) in 16-31.
    // All 16 columns identical -> scatter from lanes r and r+16 only.
#pragma unroll
    for (int r = 0; r < 8; ++r) {
      if (lane == r)      { pp_sh[m0 + r]      = acc0[r]; pp_sh[m0 + 16 + r] = acc1[r]; }
      if (lane == r + 16) { pp_sh[m0 + 8 + r]  = acc0[r]; pp_sh[m0 + 24 + r] = acc1[r]; }
    }
    __syncthreads();   // (A) p_prior visible

    // ---- observation multiply + normalizer reduction ----
    float x  = pp_sh[tid] * obs;
    float ws = wave_sum(x);
    if (lane == 0) atomicAdd(&sacc[t & 1], ws);
    __syncthreads();   // (B) sum complete

    float s   = sacc[t & 1];
    float st  = x * __builtin_amdgcn_rcpf(s);
    u_sh[tid] = (__bf16)st;                   // feed next step

    const bool writing = (t >= t_write);
    if (writing) {
      __builtin_nontemporal_store(st, &outP[t * DIM + tid]);  // streamed, never re-read
      float e  = -st * __logf(st + 1e-10f);
      float we = wave_sum(e);
      if (lane == 0) atomicAdd(&eacc[t & 1], we);
    }
    __syncthreads();   // (C) u_sh + entropy complete

    if (tid == 0) {
      if (writing) __builtin_nontemporal_store(eacc[t & 1], &outU[t]);
      eacc[t & 1] = 0.0f;
      sacc[t & 1] = 0.0f;
    }
  }
}

extern "C" void kernel_launch(void* const* d_in, const int* in_sizes, int n_in,
                              void* d_out, int out_size, void* d_ws, size_t ws_size,
                              hipStream_t stream) {
  const float* probs = (const float*)d_in[0];           // (SEQ, DIM) f32
  const float* T     = (const float*)d_in[1];           // (DIM, DIM) f32
  float* outP = (float*)d_out;                          // (SEQ, DIM) f32
  float* outU = outP + (size_t)SEQ * DIM;               // (SEQ,) f32
  hipLaunchKernelGGL(prior_layer_kernel, dim3(NCHUNKS), dim3(256), 0, stream,
                     probs, T, outP, outU);
}
